// RelationNetwork_12197707121395
// MI455X (gfx1250) — matrix operand order repaired and checked by reading, compiled, not run on hardware
//
#include <hip/hip_runtime.h>
#include <hip/hip_bf16.h>

#define AS1 __attribute__((address_space(1)))
#define AS3 __attribute__((address_space(3)))

typedef __bf16 bf16_t;
typedef __attribute__((ext_vector_type(16))) __bf16 v16bf;
typedef __attribute__((ext_vector_type(8)))  float  v8f;
typedef __attribute__((ext_vector_type(4)))  float  vf4;

#define MTOT   32767    // N-1 pair rows
#define NOBJ   32768
#define DOBJ   256
#define HGD    1024
#define TM     32       // rows per workgroup block
#define NTHR   512      // 16 waves
#define BSROWB 80       // padded LDS row stride for staged B (64B data + 16B pad)
#define LDBE   40       // BSROWB in bf16 elements
#define BSBUF  (HGD * BSROWB)   // 80 KB per B buffer

static __device__ __forceinline__ bf16_t f2bf(float f) {
  unsigned u = __float_as_uint(f);
  u += 0x7FFFu + ((u >> 16) & 1u);          // round-to-nearest-even
  union { unsigned short s; bf16_t b; } cv;
  cv.s = (unsigned short)(u >> 16);
  return cv.b;
}
static __device__ __forceinline__ float bf2f(bf16_t b) {
  union { bf16_t b; unsigned short s; } cv; cv.b = b;
  return __uint_as_float(((unsigned)cv.s) << 16);
}

union Frag { v16bf v; vf4 f4[2]; };

// ---- async-DMA stage of one B slab: Wt[0..1023][k0..k0+31] -> LDS buffer ----
// 512 threads, 4 per row: 8 x global_load_async_to_lds_b128 per thread.
static __device__ __forceinline__ void stage_slab(unsigned long long wt_b, int K,
                                                  unsigned bs, int tid)
{
  const int c  = tid & 3;        // 16B chunk within row
  const int nb = tid >> 2;       // base row (0..127)
#pragma unroll
  for (int r = 0; r < 8; ++r) {
    const int n = nb + (r << 7);
    const unsigned voff = (unsigned)(n * (K << 1) + (c << 4));    // global byte off
    const unsigned lds  = bs + (unsigned)(n * BSROWB + (c << 4)); // LDS byte addr
    asm volatile("global_load_async_to_lds_b128 %0, %1, %2"
                 :: "v"(lds), "v"(voff), "s"(wt_b) : "memory");
  }
}

// ---- one fused layer: C(TM x 1024) = relu(A(TM x K) @ W + bias) -------------
// A, C in LDS (AS3). W streamed from global via double-buffered async LDS slabs.
static __device__ void gemm_layer(const AS3 bf16_t* A, int K,
                                  unsigned long long wt_b,
                                  const AS1 float* bias,
                                  AS3 bf16_t* C,
                                  unsigned bs0, unsigned bs1)
{
  const int tid   = threadIdx.x;
  const int lane  = tid & 31;
  const int wave  = tid >> 5;         // 16 waves
  const int lhalf = lane >> 4;
  const int l15   = lane & 15;
  const int m0 = (wave & 1) * 16;     // wave's M tile
  const int n0 = (wave >> 1) * 128;   // wave's 128-col strip (8 n-tiles)

  v8f acc[8] = {};

  const int nslab = K >> 5;
  stage_slab(wt_b, K, bs0, tid);                       // prologue: slab 0
  for (int s = 0; s < nslab; ++s) {
    const unsigned bcur = (s & 1) ? bs1 : bs0;
    if (s + 1 < nslab) {                               // prefetch next slab
      stage_slab(wt_b + (unsigned long long)((s + 1) << 6), K,
                 (s & 1) ? bs0 : bs1, tid);
      asm volatile("s_wait_asynccnt 8" ::: "memory");  // my slab-s loads landed
    } else {
      asm volatile("s_wait_asynccnt 0" ::: "memory");
    }
    __syncthreads();                                   // everyone's slab s landed

    const int k0 = s << 5;
    Frag a;                                            // A frag 16x32 from LDS
    const AS3 bf16_t* abase = A + (m0 + l15) * K + k0 + (lhalf << 4);
    a.f4[0] = *(const AS3 vf4*)abase;
    a.f4[1] = *(const AS3 vf4*)(abase + 8);

    const AS3 bf16_t* B = (const AS3 bf16_t*)bcur;
    Frag bf[2];                                        // SW-pipelined B frags
    {
      const AS3 bf16_t* bb = B + (n0 + l15) * LDBE + (lhalf << 4);
      bf[0].f4[0] = *(const AS3 vf4*)bb;
      bf[0].f4[1] = *(const AS3 vf4*)(bb + 8);
    }
#pragma unroll
    for (int t = 0; t < 8; ++t) {
      if (t < 7) {                                     // load t+1 before wmma t
        const AS3 bf16_t* bb = B + (n0 + (t + 1) * 16 + l15) * LDBE + (lhalf << 4);
        bf[(t + 1) & 1].f4[0] = *(const AS3 vf4*)bb;
        bf[(t + 1) & 1].f4[1] = *(const AS3 vf4*)(bb + 8);
      }
      acc[t] = __builtin_amdgcn_wmma_f32_16x16x32_bf16(
          false, a.v, false, bf[t & 1].v, (short)0, acc[t], false, false);
    }
    __syncthreads();                                   // done reading slab s
  }

  // epilogue: bias + ReLU + bf16 -> LDS
  const int mrow0 = m0 + (lhalf << 3);
#pragma unroll
  for (int t = 0; t < 8; ++t) {
    const int n  = n0 + t * 16 + l15;
    const float bv = bias[n];
#pragma unroll
    for (int j = 0; j < 8; ++j) {
      float v = acc[t][j] + bv;
      v = fmaxf(v, 0.0f);
      C[(mrow0 + j) * HGD + n] = f2bf(v);
    }
  }
}

// ---- fused 3-layer g-MLP + masked row-sum --------------------------------
__global__ void __launch_bounds__(NTHR) k_rn_g(
    const float* x,
    const bf16_t* Wt0, const float* b0p,
    const bf16_t* Wt1, const float* b1,
    const bf16_t* Wt2, const float* b2,
    float* emb)
{
  extern __shared__ __align__(16) char smem_raw[];
  const unsigned lds0 = (unsigned)(unsigned long long)(void*)smem_raw;
  AS3 bf16_t* R0 = (AS3 bf16_t*)lds0;                        // 64 KB
  AS3 bf16_t* R1 = (AS3 bf16_t*)(lds0 + (unsigned)(TM * HGD * 2));
  const unsigned bs0 = lds0 + (unsigned)(2 * TM * HGD * 2);  // 2 x 80 KB B bufs
  const unsigned bs1 = bs0 + (unsigned)BSBUF;

  const int r0 = blockIdx.x * TM;
  const AS1 float* xg = (const AS1 float*)(unsigned long long)x;

  // stage A0 (TM x 512): [x[r], x[r+1]] along K, fp32 -> bf16
  for (int idx = threadIdx.x; idx < TM * 512; idx += NTHR) {
    const int m = idx >> 9, k = idx & 511;
    const int gr  = r0 + m + (k >> 8);
    const int col = k & 255;
    float v = (gr < NOBJ) ? xg[(size_t)gr * DOBJ + col] : 0.0f;
    R0[m * 512 + k] = f2bf(v);
  }
  __syncthreads();

  gemm_layer(R0, 512,  (unsigned long long)Wt0,
             (const AS1 float*)(unsigned long long)b0p, R1, bs0, bs1);
  __syncthreads();
  gemm_layer(R1, 1024, (unsigned long long)Wt1,
             (const AS1 float*)(unsigned long long)b1,  R0, bs0, bs1);
  __syncthreads();
  gemm_layer(R0, 1024, (unsigned long long)Wt2,
             (const AS1 float*)(unsigned long long)b2,  R1, bs0, bs1);
  __syncthreads();

  // masked column sums -> fp32 atomic accumulate into embedding
  for (int c = threadIdx.x; c < HGD; c += NTHR) {
    float s = 0.0f;
    for (int m = 0; m < TM; ++m)
      if (r0 + m < MTOT) s += bf2f(R1[m * HGD + c]);
    unsafeAtomicAdd(&emb[c], s);
  }
}

// ---- prep: zero emb + fold q @ W0[512:768] into bias0 --------------------
__global__ void k_prep(const float* q, const float* gW0, const float* gb0,
                       float* emb, float* b0p)
{
  const int n = blockIdx.x * 256 + threadIdx.x;
  if (n < HGD) {
    emb[n] = 0.0f;
    float s = gb0[n];
    for (int j = 0; j < DOBJ; ++j)
      s += q[j] * gW0[(size_t)(512 + j) * HGD + n];
    b0p[n] = s;
  }
}

// ---- weight transpose + fp32->bf16 (K x N -> N x K) ----------------------
__global__ void k_cvt_w(const float* src, bf16_t* dst, int K, int N)
{
  const int idx = blockIdx.x * 256 + threadIdx.x;
  if (idx < K * N) {
    const int k = idx / N;
    const int n = idx - k * N;
    dst[(size_t)n * K + k] = f2bf(src[idx]);
  }
}

// ---- tiny f-MLP: 1024->1024->512->256, single block ----------------------
__global__ void __launch_bounds__(256) k_f(
    const float* emb,
    const float* fW0, const float* fb0,
    const float* fW1, const float* fb1,
    const float* fW2, const float* fb2,
    float* out)
{
  __shared__ float e[1024];
  __shared__ float h0[1024];
  __shared__ float h1[512];
  const int t = threadIdx.x;
  for (int i = t; i < 1024; i += 256) e[i] = emb[i];
  __syncthreads();
  for (int n = t; n < 1024; n += 256) {
    float s = fb0[n];
    for (int k = 0; k < 1024; ++k) s += e[k] * fW0[(size_t)k * 1024 + n];
    h0[n] = fmaxf(s, 0.0f);
  }
  __syncthreads();
  for (int n = t; n < 512; n += 256) {
    float s = fb1[n];
    for (int k = 0; k < 1024; ++k) s += h0[k] * fW1[(size_t)k * 512 + n];
    h1[n] = fmaxf(s, 0.0f);
  }
  __syncthreads();
  if (t < 256) {
    float s = fb2[t];
    for (int k = 0; k < 512; ++k) s += h1[k] * fW2[(size_t)k * 256 + t];
    out[t] = s;
  }
}

// ---- host entry ----------------------------------------------------------
extern "C" void kernel_launch(void* const* d_in, const int* in_sizes, int n_in,
                              void* d_out, int out_size, void* d_ws, size_t ws_size,
                              hipStream_t stream)
{
  const float* x   = (const float*)d_in[0];
  const float* q   = (const float*)d_in[1];
  const float* gW0 = (const float*)d_in[2];
  const float* gb0 = (const float*)d_in[3];
  const float* gW1 = (const float*)d_in[4];
  const float* gb1 = (const float*)d_in[5];
  const float* gW2 = (const float*)d_in[6];
  const float* gb2 = (const float*)d_in[7];
  const float* fW0 = (const float*)d_in[8];
  const float* fb0 = (const float*)d_in[9];
  const float* fW1 = (const float*)d_in[10];
  const float* fb1 = (const float*)d_in[11];
  const float* fW2 = (const float*)d_in[12];
  const float* fb2 = (const float*)d_in[13];
  float* out = (float*)d_out;

  char* ws = (char*)d_ws;
  float*  emb = (float*)ws;                                   // 4 KB
  float*  b0p = (float*)(ws + 4096);                          // 4 KB
  bf16_t* Wt0 = (bf16_t*)(ws + 8192);                         // 1024 x 512 bf16
  bf16_t* Wt1 = (bf16_t*)((char*)Wt0 + (size_t)1024 * 512  * 2);
  bf16_t* Wt2 = (bf16_t*)((char*)Wt1 + (size_t)1024 * 1024 * 2);

  k_prep<<<4, 256, 0, stream>>>(q, gW0, gb0, emb, b0p);
  k_cvt_w<<<(512  * 1024 + 255) / 256, 256, 0, stream>>>(gW0, Wt0, 512,  1024);
  k_cvt_w<<<(1024 * 1024 + 255) / 256, 256, 0, stream>>>(gW1, Wt1, 1024, 1024);
  k_cvt_w<<<(1024 * 1024 + 255) / 256, 256, 0, stream>>>(gW2, Wt2, 1024, 1024);

  const size_t shmem = (size_t)(2 * TM * HGD * 2) + 2 * BSBUF; // 288 KB LDS
  (void)hipFuncSetAttribute(reinterpret_cast<const void*>(k_rn_g),
                            hipFuncAttributeMaxDynamicSharedMemorySize, (int)shmem);
  k_rn_g<<<(MTOT + TM - 1) / TM, NTHR, shmem, stream>>>(
      x, Wt0, b0p, Wt1, gb1, Wt2, gb2, emb);

  k_f<<<1, 256, 0, stream>>>(emb, fW0, fb0, fW1, fb1, fW2, fb2, out);
}
